// DecoderLSTM_15341623181540
// MI455X (gfx1250) — matrix operand order repaired and checked by reading, not compiled, not run on hardware
//
#include <hip/hip_runtime.h>
#include <hip/hip_bf16.h>
#include <cmath>

typedef _Float16 v16h __attribute__((ext_vector_type(16)));
typedef float    v8f  __attribute__((ext_vector_type(8)));
typedef float    f4   __attribute__((ext_vector_type(4)));

#define DEV __device__ __forceinline__

// Problem dimensions (fixed by the reference setup_inputs()).
constexpr int B    = 256;
constexpr int FD   = 512;   // feature dim F
constexpr int P    = 196;   // encoder positions
constexpr int H    = 512;
constexpr int E    = 512;
constexpr int V    = 10000;
constexpr int AD   = 196;   // attention hidden A
constexpr int APAD = 208;   // A padded to multiple of 16
constexpr int T    = 20;    // decode steps = lengths-1
constexpr int KX   = E + FD + H;  // 1536 : [emb | z | h] concat
constexpr int G4   = 4 * H;       // 2048

// ---------------------------------------------------------------------------
// WMMA fragment loaders (layouts per CDNA5 ISA 7.12.2, 16-bit 16x16x32)
// ---------------------------------------------------------------------------
// A 16x32 f16: lane L -> row m = L&15; slot j -> K = (j>>3)*16 + (L>>4)*8 + (j&7)
// => two contiguous 8-halfword (16B) chunks per lane.
DEV v16h load_a(const _Float16* __restrict__ A, int lda, int m0, int k0, int lane) {
  int l = lane & 15, hi = lane >> 4;
  const _Float16* p = A + (size_t)(m0 + l) * lda + k0 + hi * 8;
  v16h r;
  ((f4*)&r)[0] = *(const f4*)(p);
  ((f4*)&r)[1] = *(const f4*)(p + 16);
  return r;
}

// B 32x16 f16 pre-packed: tile-major [kt][nt][lane][16 slots] -> each lane reads
// 32 contiguous bytes (fully coalesced: 1KB per wave per tile).
DEV v16h load_b(const _Float16* __restrict__ Bp, int Ntiles, int kt, int nt, int lane) {
  const _Float16* p = Bp + (((size_t)kt * Ntiles + nt) * 32 + lane) * 16;
  v16h r;
  ((f4*)&r)[0] = *(const f4*)(p);
  ((f4*)&r)[1] = *(const f4*)(p + 8);
  return r;
}

DEV float wave_reduce(float s) {
#pragma unroll
  for (int off = 16; off > 0; off >>= 1) s += __shfl_xor(s, off, 32);
  return s;
}

DEV float sigmoidf(float x) { return 1.f / (1.f + __expf(-x)); }

DEV v8f wmma16(v16h a, v16h b, v8f c) {
  return __builtin_amdgcn_wmma_f32_16x16x32_f16(false, a, false, b, (short)0, c, false, false);
}

// ---------------------------------------------------------------------------
// Generic f16 WMMA GEMM body, specialized at compile time on TWO (whether this
// wave owns one or two N-tiles).  Ping-pong double buffering, unrolled x2, with
// the last k-pair peeled: the steady-state loop has no conditionals, loads
// write their registers directly, and each WMMA pair waits only for its own
// fragment set while the other set's loads are in flight.
// ---------------------------------------------------------------------------
template <bool TWO>
DEV void gemm_body(const _Float16* __restrict__ A, const _Float16* __restrict__ Bp,
                   const float* __restrict__ bias, float* __restrict__ C,
                   int Ntiles, int K, int lda, long ldc, int act,
                   int m0, int nt0, int lane) {
  v8f acc0 = {}, acc1 = {};

  v16h a0  = load_a(A, lda, m0, 0, lane);
  v16h b00 = load_b(Bp, Ntiles, 0, nt0, lane);
  v16h b10;
  if (TWO) b10 = load_b(Bp, Ntiles, 0, nt0 + 1, lane);

  int Kt = K >> 5;                 // always even and >= 2 at every call site
  int kt = 0;
  for (; kt + 2 < Kt; kt += 2) {
    v16h a1  = load_a(A, lda, m0, (kt + 1) << 5, lane);
    v16h b01 = load_b(Bp, Ntiles, kt + 1, nt0, lane);
    v16h b11;
    if (TWO) b11 = load_b(Bp, Ntiles, kt + 1, nt0 + 1, lane);

    acc0 = wmma16(a0, b00, acc0);
    if (TWO) acc1 = wmma16(a0, b10, acc1);

    a0  = load_a(A, lda, m0, (kt + 2) << 5, lane);
    b00 = load_b(Bp, Ntiles, kt + 2, nt0, lane);
    if (TWO) b10 = load_b(Bp, Ntiles, kt + 2, nt0 + 1, lane);

    acc0 = wmma16(a1, b01, acc0);
    if (TWO) acc1 = wmma16(a1, b11, acc1);
  }
  {                                 // epilogue: kt == Kt-2, no more prefetch
    v16h a1  = load_a(A, lda, m0, (kt + 1) << 5, lane);
    v16h b01 = load_b(Bp, Ntiles, kt + 1, nt0, lane);
    v16h b11;
    if (TWO) b11 = load_b(Bp, Ntiles, kt + 1, nt0 + 1, lane);

    acc0 = wmma16(a0, b00, acc0);
    if (TWO) acc1 = wmma16(a0, b10, acc1);
    acc0 = wmma16(a1, b01, acc0);
    if (TWO) acc1 = wmma16(a1, b11, acc1);
  }

  // C 16x16 f32: lanes 0-15 -> N=lane, M=r ; lanes 16-31 -> N=lane-16, M=r+8
  int l = lane & 15, hi = lane >> 4;
  int n0 = nt0 << 4;
  float bs0 = bias ? bias[n0 + l] : 0.f;
  float bs1 = 0.f;
  if (TWO) bs1 = bias ? bias[n0 + 16 + l] : 0.f;
#pragma unroll
  for (int r = 0; r < 8; ++r) {
    int m = m0 + r + hi * 8;
    float v0 = acc0[r] + bs0;
    if (act == 1) v0 = sigmoidf(v0);
    C[(size_t)m * ldc + n0 + l] = v0;
    if (TWO) {
      float v1 = acc1[r] + bs1;
      if (act == 1) v1 = sigmoidf(v1);
      C[(size_t)m * ldc + n0 + 16 + l] = v1;
    }
  }
}

// Block = 8 waves covering 8 consecutive M-tiles of ONE N-pair:
//   - N-pair index comes from blockIdx.x (scalar) -> TWO dispatch is one SGPR branch
//   - M-tile index = wave id                      -> no guard (Mtiles % 8 == 0)
__global__ void __launch_bounds__(256)
wmma_gemm(const _Float16* __restrict__ A, const _Float16* __restrict__ Bp,
          const float* __restrict__ bias, float* __restrict__ C,
          int nPair, int Ntiles, int K, int lda, long ldc, int act) {
  int lane = threadIdx.x & 31;
  int wave = threadIdx.x >> 5;
  int np   = blockIdx.x % nPair;              // scalar
  int mt   = (blockIdx.x / nPair) * 8 + wave; // wave-varying (addressing only)
  int m0   = mt << 4;
  int nt0  = np * 2;
  if (nt0 + 1 < Ntiles)
    gemm_body<true >(A, Bp, bias, C, Ntiles, K, lda, ldc, act, m0, nt0, lane);
  else
    gemm_body<false>(A, Bp, bias, C, Ntiles, K, lda, ldc, act, m0, nt0, lane);
}

// ---------------------------------------------------------------------------
// Weight packing: f32 row-major [K x N] (optionally two stacked sources) ->
// f16 WMMA-B fragment order, zero-padded to Ntiles*16 columns.
// ---------------------------------------------------------------------------
__global__ void pack_b(const float* __restrict__ srcA, int rowsA,
                       const float* __restrict__ srcB, _Float16* __restrict__ dst,
                       int N, int ldsrc, int Ntiles, int Ktiles) {
  size_t idx = (size_t)blockIdx.x * 256 + threadIdx.x;
  size_t total = (size_t)Ktiles * Ntiles * 512;
  if (idx >= total) return;
  int j = (int)(idx & 15);
  int L = (int)((idx >> 4) & 31);
  size_t tile = idx >> 9;
  int nt = (int)(tile % Ntiles);
  int kt = (int)(tile / Ntiles);
  int k = kt * 32 + ((j >> 3) << 4) + ((L >> 4) << 3) + (j & 7);   // slot j -> K = (j>>3)*16 + (L>>4)*8 + (j&7) within tile
  int n = (nt << 4) + (L & 15);
  float v = 0.f;
  if (n < N)
    v = (k < rowsA) ? srcA[(size_t)k * ldsrc + n] : srcB[(size_t)(k - rowsA) * ldsrc + n];
  dst[idx] = (_Float16)v;
}

__global__ void cvt_f16(const float* __restrict__ src, _Float16* __restrict__ dst, size_t n) {
  size_t i = (size_t)blockIdx.x * 256 + threadIdx.x;
  if (i < n) dst[i] = (_Float16)src[i];
}

__global__ void prep_small(const float* b1, const float* w2, const float* bih, const float* bhh,
                           float* b1pad, float* w2pad, float* bcat) {
  int i = blockIdx.x * 256 + threadIdx.x;
  if (i < APAD) {
    b1pad[i] = (i < AD) ? b1[i] : 0.f;
    w2pad[i] = (i < AD) ? w2[i] : 0.f;
  }
  if (i < G4) bcat[i] = bih[i] + bhh[i];
}

__global__ void init_state(_Float16* hbuf, _Float16* xcat, float* c) {
  int i = blockIdx.x * 256 + threadIdx.x;
  if (i >= B * H) return;
  int b = i >> 9, j = i & 511;
  hbuf[i] = (_Float16)0.f;
  xcat[b * KX + (E + FD) + j] = (_Float16)0.f;
  c[i] = 0.f;
}

__global__ void emb_gather(const float* __restrict__ table, const int* __restrict__ caps,
                           _Float16* __restrict__ xcat, int t) {
  int i = blockIdx.x * 256 + threadIdx.x;
  if (i >= B * E) return;
  int b = i >> 9, e = i & 511;
  int tok = caps[b * (T + 1) + t];
  xcat[b * KX + e] = (_Float16)table[(size_t)tok * E + e];
}

// score[b][p] = att_b2 + sum_a tanh(pre1[p*B+b, a] + hproj[b, a]) * w2[a]
// (padded cols 196..207 are exact zeros on all three operands)
__global__ void score_kernel(const float* __restrict__ pre1, const float* __restrict__ hproj,
                             const float* __restrict__ w2pad, const float* __restrict__ b2,
                             float* __restrict__ scores) {
  int wid  = blockIdx.x * 8 + (threadIdx.x >> 5);
  int lane = threadIdx.x & 31;
  if (wid >= B * P) return;
  int b = wid & 255, p = wid >> 8;
  const float* pr = pre1 + (size_t)(p * B + b) * APAD;
  const float* hr = hproj + b * APAD;
  float s = 0.f;
  for (int a = lane; a < APAD; a += 32)
    s += tanhf(pr[a] + hr[a]) * w2pad[a];
  s = wave_reduce(s);
  if (lane == 0) scores[b * P + p] = s + b2[0];
}

__global__ void softmax_kernel(const float* __restrict__ scores, float* __restrict__ alpha,
                               float* __restrict__ alphas_out, int t) {
  __shared__ float red[256];
  int b = blockIdx.x, tid = threadIdx.x;
  float v = (tid < P) ? scores[b * P + tid] : -3.4e38f;
  red[tid] = v; __syncthreads();
  for (int s = 128; s > 0; s >>= 1) { if (tid < s) red[tid] = fmaxf(red[tid], red[tid + s]); __syncthreads(); }
  float mx = red[0]; __syncthreads();
  float e = (tid < P) ? __expf(v - mx) : 0.f;
  red[tid] = e; __syncthreads();
  for (int s = 128; s > 0; s >>= 1) { if (tid < s) red[tid] += red[tid + s]; __syncthreads(); }
  float inv = 1.f / red[0];
  if (tid < P) {
    float al = e * inv;
    alpha[b * P + tid] = al;
    alphas_out[(size_t)b * T * P + (size_t)t * P + tid] = al;
  }
}

// z[b][f] = (sum_p features[b][f][p] * alpha[b][p]) * beta[b][f]  -> xcat z slot (f16)
__global__ void zgate_kernel(const float* __restrict__ features, const float* __restrict__ alpha,
                             const float* __restrict__ beta, _Float16* __restrict__ xcat) {
  int wid  = blockIdx.x * 8 + (threadIdx.x >> 5);
  int lane = threadIdx.x & 31;
  if (wid >= B * FD) return;
  int b = wid >> 9, f = wid & 511;
  const float* fr = features + (size_t)wid * P;   // features[b][f][:] contiguous
  const float* al = alpha + b * P;
  float s = 0.f;
  for (int a = lane; a < P; a += 32) s += fr[a] * al[a];
  s = wave_reduce(s);
  if (lane == 0) xcat[b * KX + E + f] = (_Float16)(s * beta[wid]);
}

__global__ void lstm_cell(const float* __restrict__ gates, float* __restrict__ c,
                          _Float16* __restrict__ hbuf, _Float16* __restrict__ xcat) {
  int i = blockIdx.x * 256 + threadIdx.x;
  if (i >= B * H) return;
  int b = i >> 9, j = i & 511;
  const float* g = gates + (size_t)b * G4;
  float ig = sigmoidf(g[j]);
  float fg = sigmoidf(g[H + j]);
  float gg = tanhf(g[2 * H + j]);
  float og = sigmoidf(g[3 * H + j]);
  float cn = fg * c[i] + ig * gg;
  float hn = og * tanhf(cn);
  c[i] = cn;
  _Float16 hh = (_Float16)hn;
  hbuf[i] = hh;
  xcat[b * KX + E + FD + j] = hh;
}

// ---------------------------------------------------------------------------
extern "C" void kernel_launch(void* const* d_in, const int* in_sizes, int n_in,
                              void* d_out, int out_size, void* d_ws, size_t ws_size,
                              hipStream_t stream) {
  const float* features    = (const float*)d_in[0];
  const int*   captions    = (const int*)  d_in[1];
  // d_in[2] = lengths (static 21 -> T=20)
  const float* embed_table = (const float*)d_in[3];
  const float* att_w1      = (const float*)d_in[4];
  const float* att_b1      = (const float*)d_in[5];
  const float* att_w2      = (const float*)d_in[6];
  const float* att_b2      = (const float*)d_in[7];
  const float* beta_w      = (const float*)d_in[8];
  const float* beta_b      = (const float*)d_in[9];
  const float* w_ih        = (const float*)d_in[10];
  const float* w_hh        = (const float*)d_in[11];
  const float* b_ih        = (const float*)d_in[12];
  const float* b_hh        = (const float*)d_in[13];
  const float* out_w       = (const float*)d_in[14];
  const float* out_b       = (const float*)d_in[15];

  float* pred       = (float*)d_out;                       // [B,T,V]
  float* alphas_out = pred + (size_t)B * T * V;            // [B,T,P]

  // --- workspace bump allocator (256B aligned) ---
  char* wp = (char*)d_ws;
  auto alloc = [&](size_t bytes) -> void* {
    void* p = (void*)wp;
    wp += (bytes + 255) & ~(size_t)255;
    return p;
  };
  _Float16* feats16 = (_Float16*)alloc((size_t)B * FD * P * 2);     // A of pre1 GEMM
  _Float16* w1f_p   = (_Float16*)alloc((size_t)16 * 13 * 512 * 2);  // att_w1[:F]  packed
  _Float16* w1h_p   = (_Float16*)alloc((size_t)16 * 13 * 512 * 2);  // att_w1[F:]  packed
  _Float16* beta_p  = (_Float16*)alloc((size_t)16 * 32 * 512 * 2);  // beta_w      packed
  _Float16* wcat_p  = (_Float16*)alloc((size_t)48 * 128 * 512 * 2); // [w_ih;w_hh] packed
  _Float16* outw_p  = (_Float16*)alloc((size_t)16 * 625 * 512 * 2); // out_w       packed
  float* pre1   = (float*)alloc((size_t)P * B * APAD * 4);          // hoisted attention term
  float* hproj  = (float*)alloc((size_t)B * APAD * 4);
  float* scores = (float*)alloc((size_t)B * P * 4);
  float* alpha  = (float*)alloc((size_t)B * P * 4);
  float* betab  = (float*)alloc((size_t)B * FD * 4);
  float* gates  = (float*)alloc((size_t)B * G4 * 4);
  float* cbuf   = (float*)alloc((size_t)B * H * 4);
  _Float16* hbuf16 = (_Float16*)alloc((size_t)B * H * 2);
  _Float16* xcat   = (_Float16*)alloc((size_t)B * KX * 2);          // [emb | z | h] f16
  float* b1pad = (float*)alloc(APAD * 4);
  float* w2pad = (float*)alloc(APAD * 4);
  float* bcat  = (float*)alloc(G4 * 4);

  // Mtiles must be a multiple of 8 (block = 8 waves stacked along M).
  auto gemm = [&](const _Float16* A, const _Float16* Bp, const float* bias, float* C,
                  int Mtiles, int Ntiles, int K, int lda, long ldc, int act) {
    int nPair = (Ntiles + 1) / 2;
    int blocks = (Mtiles / 8) * nPair;
    wmma_gemm<<<blocks, 256, 0, stream>>>(A, Bp, bias, C, nPair, Ntiles, K, lda, ldc, act);
  };
  auto pack = [&](const float* srcA, int rowsA, const float* srcB, _Float16* dst,
                  int N, int ldsrc, int Ntiles, int Ktiles) {
    size_t total = (size_t)Ktiles * Ntiles * 512;
    pack_b<<<(unsigned)((total + 255) / 256), 256, 0, stream>>>(srcA, rowsA, srcB, dst,
                                                               N, ldsrc, Ntiles, Ktiles);
  };

  // ---- one-time setup ----
  {
    size_t nf = (size_t)B * FD * P;
    cvt_f16<<<(unsigned)((nf + 255) / 256), 256, 0, stream>>>(features, feats16, nf);
  }
  pack(att_w1,                  512, att_w1, w1f_p, AD, AD, 13, 16);
  pack(att_w1 + (size_t)512*AD, 512, att_w1, w1h_p, AD, AD, 13, 16);
  pack(beta_w, 512, beta_w, beta_p, FD, FD, 32, 16);
  pack(w_ih,  1024, w_hh,  wcat_p, G4, G4, 128, 48);
  pack(out_w,  512, out_w, outw_p, V,  V,  625, 16);
  prep_small<<<8, 256, 0, stream>>>(att_b1, att_w2, b_ih, b_hh, b1pad, w2pad, bcat);
  init_state<<<512, 256, 0, stream>>>(hbuf16, xcat, cbuf);
  // pre1[P*B, 208] = feats_flat @ att_w1[:F] + att_b1  (time-invariant, done once)
  gemm(feats16, w1f_p, b1pad, pre1, (P * B) / 16, 13, 512, FD, APAD, 0);

  // ---- recurrent decode: T steps, all kernels serialized on `stream` ----
  for (int t = 0; t < T; ++t) {
    emb_gather<<<512, 256, 0, stream>>>(embed_table, captions, xcat, t);
    // hproj[B,208] = h @ att_w1[F:]
    gemm(hbuf16, w1h_p, nullptr, hproj, B / 16, 13, 512, H, APAD, 0);
    score_kernel<<<(B * P) / 8, 256, 0, stream>>>(pre1, hproj, w2pad, att_b2, scores);
    softmax_kernel<<<B, 256, 0, stream>>>(scores, alpha, alphas_out, t);
    // beta[B,F] = sigmoid(h @ beta_w + beta_b)   (sigmoid fused)
    gemm(hbuf16, beta_p, beta_b, betab, B / 16, 32, 512, H, FD, 1);
    zgate_kernel<<<(B * FD) / 8, 256, 0, stream>>>(features, alpha, betab, xcat);
    // gates[B,2048] = [emb|z|h] @ [w_ih;w_hh] + (b_ih+b_hh)
    gemm(xcat, wcat_p, bcat, gates, B / 16, 128, KX, KX, G4, 0);
    lstm_cell<<<512, 256, 0, stream>>>(gates, cbuf, hbuf16, xcat);
    // logits -> predictions[:, t, :]   (ldc = T*V row stride)
    gemm(hbuf16, outw_p, out_b, pred + (size_t)t * V, B / 16, 625, 512, H, (long)T * V, 0);
  }
}
